// simpleHeteroGATConv_39745627357804
// MI455X (gfx1250) — compile-verified, matrix operand order checked by reading, not run on hardware
//
#include <hip/hip_runtime.h>

#define NNODES 50000
#define NEDGES 800000
#define INF_   256
#define HEADS  4
#define FOUT   64
#define EFEAT  64
#define NTYPE  8
#define HF     256   // HEADS*FOUT == IN

typedef __attribute__((ext_vector_type(16))) __bf16 v16bf;
typedef __attribute__((ext_vector_type(8)))  float  v8f;

union Frag { v16bf v; unsigned short s[16]; uint4 q[2]; };

__device__ __forceinline__ unsigned short f32_to_bf16(float x) {
  unsigned u = __float_as_uint(x);
  unsigned r = 0x7FFFu + ((u >> 16) & 1u);
  return (unsigned short)((u + r) >> 16);
}
__device__ __forceinline__ unsigned f2ord(float x) {
  unsigned u = __float_as_uint(x);
  return (u & 0x80000000u) ? ~u : (u | 0x80000000u);
}
__device__ __forceinline__ float ord2f(unsigned u) {
  return __uint_as_float((u & 0x80000000u) ? (u & 0x7FFFFFFFu) : ~u);
}

// ---- f32 -> bf16 convert (vectorized x4) ----
__global__ void k_cvt_bf16(const float* __restrict__ x, unsigned short* __restrict__ y, int n4) {
  int i = blockIdx.x * blockDim.x + threadIdx.x;
  if (i >= n4) return;
  float4 v = ((const float4*)x)[i];
  ushort4 o;
  o.x = f32_to_bf16(v.x); o.y = f32_to_bf16(v.y);
  o.z = f32_to_bf16(v.z); o.w = f32_to_bf16(v.w);
  ((ushort4*)y)[i] = o;
}

// ---- pack fc_w[256,256] into per-lane WMMA B fragments (bf16) ----
// layout: [kt(8)][ntile(16)][lane(32)][16 bf16], k = kt*32 + half*8 + (i<8 ? i : i+8)
__global__ void k_pack_b(const float* __restrict__ W, unsigned short* __restrict__ B) {
  int idx = blockIdx.x * blockDim.x + threadIdx.x;
  if (idx >= 8 * 16 * 32) return;
  int lane = idx & 31;
  int nt   = (idx >> 5) & 15;
  int kt   = idx >> 9;
  int n    = nt * 16 + (lane & 15);
  int half = lane >> 4;
  unsigned short* out = B + idx * 16;
#pragma unroll
  for (int i = 0; i < 16; ++i) {
    int k = kt * 32 + half * 8 + (i < 8 ? i : i + 8);
    out[i] = f32_to_bf16(W[k * HF + n]);
  }
}

// ---- WMMA GEMM: feat[50000,256] = nfeat_bf16 @ Bpack, one wave -> 16x64 strip ----
__global__ void k_gemm_wmma(const unsigned short* __restrict__ A,
                            const unsigned short* __restrict__ B,
                            float* __restrict__ C) {
  int wave = (blockIdx.x * blockDim.x + threadIdx.x) >> 5;
  int lane = threadIdx.x & 31;
  if (wave >= 3125 * 4) return;
  int mtile = wave >> 2;        // 0..3124
  int ng    = wave & 3;         // n-group of 4 tiles
  int half  = lane >> 4;
  int mrow  = mtile * 16 + (lane & 15);

  v8f acc[4];
#pragma unroll
  for (int j = 0; j < 4; ++j) acc[j] = (v8f){0.f,0.f,0.f,0.f,0.f,0.f,0.f,0.f};

  const unsigned short* arow = A + (size_t)mrow * HF + half * 8;

#pragma unroll
  for (int kt = 0; kt < 8; ++kt) {
    Frag af;
    af.q[0] = *(const uint4*)(arow + kt * 32);        // K = kbase + half*8 .. +7
    af.q[1] = *(const uint4*)(arow + kt * 32 + 16);   // K = kbase + 16 + half*8 ..
    if (kt < 7) __builtin_prefetch(arow + (kt + 1) * 32, 0, 0);
#pragma unroll
    for (int j = 0; j < 4; ++j) {
      int ntile = ng * 4 + j;
      const uint4* bp = (const uint4*)(B + (((size_t)kt * 16 + ntile) * 32 + lane) * 16);
      Frag bf_;
      bf_.q[0] = bp[0];
      bf_.q[1] = bp[1];
      acc[j] = __builtin_amdgcn_wmma_f32_16x16x32_bf16(
          false, af.v, false, bf_.v, (short)0, acc[j], false, false);
    }
  }

#pragma unroll
  for (int j = 0; j < 4; ++j) {
    int ncol = (ng * 4 + j) * 16 + (lane & 15);
    float* crow = C + (size_t)(mtile * 16 + 8 * half) * HF + ncol;
#pragma unroll
    for (int r = 0; r < 8; ++r) crow[(size_t)r * HF] = acc[j][r];
  }
}

// ---- per-node attention halves el/er: one thread per (n,h) ----
__global__ void k_elr(const float* __restrict__ feat,
                      const float* __restrict__ attn_l, const float* __restrict__ attn_r,
                      float* __restrict__ el, float* __restrict__ er) {
  int idx = blockIdx.x * blockDim.x + threadIdx.x;
  if (idx >= NNODES * HEADS) return;
  int n = idx >> 2, h = idx & 3;
  const float* f  = feat + (size_t)n * HF + h * FOUT;
  const float* al = attn_l + h * FOUT;
  const float* ar = attn_r + h * FOUT;
  float sl = 0.f, sr = 0.f;
#pragma unroll 4
  for (int i = 0; i < FOUT; ++i) { float v = f[i]; sl += v * al[i]; sr += v * ar[i]; }
  el[idx] = sl; er[idx] = sr;
}

// ---- per-etype edge attention ee[NT,H] (tiny: 1 block of 32) ----
__global__ void k_ee(const float* __restrict__ edge_emb, const float* __restrict__ fc_e_w,
                     const float* __restrict__ attn_e, float* __restrict__ ee) {
  int tid = threadIdx.x;
  if (tid >= NTYPE * HEADS) return;
  int t = tid >> 2, h = tid & 3;
  float s = 0.f;
  for (int e2 = 0; e2 < EFEAT; ++e2) {
    float acc = 0.f;
    for (int k = 0; k < EFEAT; ++k)
      acc += edge_emb[t * EFEAT + k] * fc_e_w[k * (HEADS * EFEAT) + h * EFEAT + e2];
    s += acc * attn_e[h * EFEAT + e2];
  }
  ee[t * HEADS + h] = s;
}

// ---- edge pass 1: raw score + leaky relu + segment max (ordered-uint atomic) ----
__global__ void k_edge1(const int* __restrict__ src, const int* __restrict__ dst,
                        const int* __restrict__ ety,
                        const float* __restrict__ el, const float* __restrict__ er,
                        const float* __restrict__ ee,
                        float* __restrict__ araw, unsigned* __restrict__ segmax) {
  int idx = blockIdx.x * blockDim.x + threadIdx.x;
  if (idx >= NEDGES * HEADS) return;
  int e = idx >> 2, h = idx & 3;
  int s = src[e], d = dst[e], t = ety[e];
  float x = el[s * HEADS + h] + er[d * HEADS + h] + ee[t * HEADS + h];
  float r = x > 0.f ? x : 0.2f * x;
  araw[idx] = r;
  atomicMax(&segmax[d * HEADS + h], f2ord(r));
}

// ---- edge pass 2: exp(x - max) + segment sum ----
__global__ void k_edge2(const int* __restrict__ dst,
                        const unsigned* __restrict__ segmax,
                        float* __restrict__ araw, float* __restrict__ segsum) {
  int idx = blockIdx.x * blockDim.x + threadIdx.x;
  if (idx >= NEDGES * HEADS) return;
  int e = idx >> 2, h = idx & 3;
  int d = dst[e];
  float m = ord2f(segmax[d * HEADS + h]);
  float v = __expf(araw[idx] - m);
  araw[idx] = v;  // becomes a_exp in place
  atomicAdd(&segsum[d * HEADS + h], v);
}

// ---- edge pass 3: normalize, emit a, scatter-accumulate rst (one wave / edge) ----
__global__ void k_agg(const float* __restrict__ feat, const float* __restrict__ aexp,
                      const float* __restrict__ segsum,
                      const int* __restrict__ src, const int* __restrict__ dst,
                      float* __restrict__ rst, float* __restrict__ a_out) {
  int wave = (blockIdx.x * blockDim.x + threadIdx.x) >> 5;
  int lane = threadIdx.x & 31;
  if (wave >= NEDGES) return;
  int s = src[wave], d = dst[wave];
  int h = lane >> 3;  // 8 lanes per head
  float av = aexp[(size_t)wave * HEADS + h] / segsum[d * HEADS + h];
  if ((lane & 7) == 0) a_out[(size_t)wave * HEADS + h] = av;
  int base = h * FOUT + (lane & 7) * 8;
  const float4* fs = (const float4*)(feat + (size_t)s * HF + base);
  float4 v0 = fs[0], v1 = fs[1];
  float* rd = rst + (size_t)d * HF + base;
  atomicAdd(rd + 0, av * v0.x);
  atomicAdd(rd + 1, av * v0.y);
  atomicAdd(rd + 2, av * v0.z);
  atomicAdd(rd + 3, av * v0.w);
  atomicAdd(rd + 4, av * v1.x);
  atomicAdd(rd + 5, av * v1.y);
  atomicAdd(rd + 6, av * v1.z);
  atomicAdd(rd + 7, av * v1.w);
}

extern "C" void kernel_launch(void* const* d_in, const int* in_sizes, int n_in,
                              void* d_out, int out_size, void* d_ws, size_t ws_size,
                              hipStream_t stream) {
  const float* nfeat    = (const float*)d_in[0];
  const float* fc_w     = (const float*)d_in[1];
  const float* fc_e_w   = (const float*)d_in[2];
  const float* attn_l   = (const float*)d_in[3];
  const float* attn_r   = (const float*)d_in[4];
  const float* attn_e   = (const float*)d_in[5];
  const float* edge_emb = (const float*)d_in[6];
  const int*   src      = (const int*)d_in[7];
  const int*   dst      = (const int*)d_in[8];
  const int*   etype    = (const int*)d_in[9];

  float* rst   = (float*)d_out;                       // [N, H*F]
  float* a_out = (float*)d_out + (size_t)NNODES * HF; // [E, H]

  // workspace layout (all 16B aligned)
  char* ws = (char*)d_ws;
  size_t o = 0;
  float*          feat  = (float*)(ws + o);          o += (size_t)NNODES * HF * 4;   // 51.2 MB
  unsigned short* nfbf  = (unsigned short*)(ws + o); o += (size_t)NNODES * HF * 2;   // 25.6 MB
  unsigned short* bpack = (unsigned short*)(ws + o); o += (size_t)8 * 16 * 32 * 16 * 2; // 128 KB
  float*          el    = (float*)(ws + o);          o += (size_t)NNODES * HEADS * 4;
  float*          er    = (float*)(ws + o);          o += (size_t)NNODES * HEADS * 4;
  float*          eev   = (float*)(ws + o);          o += 128;
  float*          araw  = (float*)(ws + o);          o += (size_t)NEDGES * HEADS * 4; // 12.8 MB
  unsigned*       smax  = (unsigned*)(ws + o);       o += (size_t)NNODES * HEADS * 4;
  float*          ssum  = (float*)(ws + o);          o += (size_t)NNODES * HEADS * 4;

  // zero accumulators (graph-capture-safe stream memsets); smax=0 encodes -inf
  hipMemsetAsync(rst,  0, (size_t)NNODES * HF * 4, stream);
  hipMemsetAsync(smax, 0, (size_t)NNODES * HEADS * 4, stream);
  hipMemsetAsync(ssum, 0, (size_t)NNODES * HEADS * 4, stream);

  // 1) convert nfeat to bf16
  {
    int n4 = NNODES * HF / 4;
    k_cvt_bf16<<<(n4 + 255) / 256, 256, 0, stream>>>(nfeat, nfbf, n4);
  }
  // 2) pack fc_w into WMMA B fragments
  k_pack_b<<<16, 256, 0, stream>>>(fc_w, bpack);
  // 3) WMMA GEMM -> feat
  {
    int waves = 3125 * 4;
    int blocks = (waves * 32 + 255) / 256;
    k_gemm_wmma<<<blocks, 256, 0, stream>>>(nfbf, bpack, feat);
  }
  // 4) el / er
  k_elr<<<(NNODES * HEADS + 255) / 256, 256, 0, stream>>>(feat, attn_l, attn_r, el, er);
  // 5) ee (tiny)
  k_ee<<<1, 32, 0, stream>>>(edge_emb, fc_e_w, attn_e, eev);
  // 6) edge pass 1: scores + segment max
  k_edge1<<<(NEDGES * HEADS + 255) / 256, 256, 0, stream>>>(src, dst, etype, el, er, eev, araw, smax);
  // 7) edge pass 2: exp + segment sum
  k_edge2<<<(NEDGES * HEADS + 255) / 256, 256, 0, stream>>>(dst, smax, araw, ssum);
  // 8) edge pass 3: normalize + scatter aggregate
  {
    int blocks = (NEDGES * 32 + 255) / 256;
    k_agg<<<blocks, 256, 0, stream>>>(feat, araw, ssum, src, dst, rst, a_out);
  }
}
